// AggLayer_4784593568488
// MI455X (gfx1250) — matrix operand order repaired and checked by reading, not compiled, not run on hardware
//
#include <hip/hip_runtime.h>
#include <hip/hip_bf16.h>

// ---------------------------------------------------------------------------
// TAGConv(K=3, in=1, out=128) + MLP(128->128->1) fused for MI455X (gfx1250).
//
// Roofline: edge streaming dominates. v2 adds an int64->packed-int32 index
// conversion pass (fused with degree accumulation) so the 3 propagation
// passes read 192MB instead of 320MB each; int32 indices (128MB) + edge_attr
// (64MB) ~fit the 192MB L2. Node arrays (2MB each) are L2-resident so the
// scatter atomics are cheap. The 16.4 GFLOP MLP layer runs entirely on
// v_wmma_f32_16x16x32_f16, fully fused per 16-node wave tile (activations
// never touch HBM).
// ---------------------------------------------------------------------------

typedef __attribute__((ext_vector_type(16))) _Float16 v16h;
typedef __attribute__((ext_vector_type(8)))  _Float16 v8h;
typedef __attribute__((ext_vector_type(8)))  float    v8f;

#define DIM 128
#define W1_STRIDE 136   // 128 + 8 halves padding -> conflict-free LDS reads

__global__ __launch_bounds__(256) void zero_f32_kernel(float* p, int n) {
    int i = blockIdx.x * 256 + threadIdx.x;
    if (i < n) p[i] = 0.0f;
}

// One-shot pass: int64 indices -> packed int2 in ws, fused degree atomics.
// Non-temporal loads: the int64 stream is read exactly once; keep L2 for the
// int32/edge_attr working set that the 3 propagation passes will re-read.
__global__ __launch_bounds__(256) void convert_deg_kernel(
    const long long* __restrict__ row64, const long long* __restrict__ col64,
    const float* __restrict__ ea, int2* __restrict__ rc,
    float* __restrict__ deg, int E) {
    int e = blockIdx.x * 256 + threadIdx.x;
    if (e >= E) return;
    int r = (int)__builtin_nontemporal_load(&row64[e]);
    int c = (int)__builtin_nontemporal_load(&col64[e]);
    rc[e] = make_int2(r, c);
    atomicAdd(&deg[c], ea[e]);
}

// Fallback (small ws): degree only, from int64 col.
__global__ __launch_bounds__(256) void deg_kernel(const long long* __restrict__ col,
                                                  const float* __restrict__ ea,
                                                  float* __restrict__ deg, int E) {
    int e = blockIdx.x * 256 + threadIdx.x;
    if (e >= E) return;
    atomicAdd(&deg[(int)col[e]], ea[e]);
}

// dis = deg > 0 ? rsqrt(deg) : 0   (in place)
__global__ __launch_bounds__(256) void dis_kernel(float* __restrict__ deg, int n) {
    int i = blockIdx.x * 256 + threadIdx.x;
    if (i >= n) return;
    float d = deg[i];
    deg[i] = (d > 0.0f) ? rsqrtf(d) : 0.0f;
}

// hout[col] += dis[row]*ea*dis[col] * hin[row]; norm recomputed on the fly
// (dis is 2MB -> L2 hit). Packed int2 indices: one global_load_b64 per edge.
__global__ __launch_bounds__(256) void prop32_kernel(const int2* __restrict__ rc,
                                                     const float* __restrict__ ea,
                                                     const float* __restrict__ dis,
                                                     const float* __restrict__ hin,
                                                     float* __restrict__ hout, int E) {
    int e = blockIdx.x * 256 + threadIdx.x;
    if (e >= E) return;
    int2 p = rc[e];
    float nv = dis[p.x] * ea[e] * dis[p.y];
    atomicAdd(&hout[p.y], nv * hin[p.x]);
}

// Fallback propagation from int64 indices.
__global__ __launch_bounds__(256) void prop64_kernel(const long long* __restrict__ row,
                                                     const long long* __restrict__ col,
                                                     const float* __restrict__ ea,
                                                     const float* __restrict__ dis,
                                                     const float* __restrict__ hin,
                                                     float* __restrict__ hout, int E) {
    int e = blockIdx.x * 256 + threadIdx.x;
    if (e >= E) return;
    int r = (int)row[e];
    int c = (int)col[e];
    float nv = dis[r] * ea[e] * dis[c];
    atomicAdd(&hout[c], nv * hin[r]);
}

// Fused node kernel: per wave tile of 16 nodes:
//   out[m,k] = relu(h0*W0[k]+h1*W1[k]+h2*W2[k]+h3*W3[k]+bias[k])  (A-fragment)
//   z = relu(out @ mlp_w1 + b1)     -> 32x v_wmma_f32_16x16x32_f16
//   y = relu(z @ mlp_w2 + b2)       -> from C-fragments + shfl reduction
__global__ __launch_bounds__(256) void node_mlp_kernel(
    const float* __restrict__ x,
    const float* __restrict__ h1a, const float* __restrict__ h2a,
    const float* __restrict__ h3a,
    const float* __restrict__ W0, const float* __restrict__ W1,
    const float* __restrict__ W2, const float* __restrict__ W3,
    const float* __restrict__ bias,
    const float* __restrict__ w1,  const float* __restrict__ b1,
    const float* __restrict__ w2,  const float* __restrict__ b2,
    float* __restrict__ out, int N) {

    __shared__ _Float16 sW1[DIM * W1_STRIDE];  // transposed: sW1[n*136 + k]
    __shared__ float    sHopT[DIM][8];         // [k][0..3]=W0..W3, [4]=bias
    __shared__ float2   sBW[DIM];              // (mlp_b1[n], mlp_w2[n])

    // Stage mlp_w1 into LDS as f16, transposed with padded stride.
    for (int i = threadIdx.x; i < DIM * DIM; i += 256) {
        int k = i >> 7, n = i & 127;           // w1 row-major [k][n]
        sW1[n * W1_STRIDE + k] = (_Float16)w1[i];
    }
    if (threadIdx.x < DIM) {
        int k = threadIdx.x;
        sHopT[k][0] = W0[k]; sHopT[k][1] = W1[k];
        sHopT[k][2] = W2[k]; sHopT[k][3] = W3[k];
        sHopT[k][4] = bias[k];
        sBW[k] = make_float2(b1[k], w2[k]);
    }
    __syncthreads();

    const int lane = threadIdx.x & 31;
    const int wave = threadIdx.x >> 5;
    const int hi   = lane >> 4;       // half-wave select
    const int m    = lane & 15;       // row within 16-node tile
    const int tile = blockIdx.x * 128 + wave * 16;
    if (tile >= N) return;            // whole-wave uniform (N % 16 == 0)

    const int node = tile + m;
    const float f0 = x[node];
    const float f1 = h1a[node];
    const float f2 = h2a[node];
    const float f3 = h3a[node];

    // Build A fragments (16x32 f16 per 32-K chunk) in contiguous K order (same as B):
// lane<16: halves 0..15 -> K 0..15; lane>=16: halves 0..15 -> K 16..31.
    // sHopT address is uniform across each 16-lane half -> LDS broadcast.
    v16h A[4];
    #pragma unroll
    for (int kc = 0; kc < 4; ++kc) {
        #pragma unroll
        for (int h = 0; h < 16; ++h) {
            int k = 32 * kc + h + (hi << 4);
            const float4 w = *(const float4*)&sHopT[k][0];  // 16B-aligned b128
            float v = f0 * w.x + f1 * w.y + f2 * w.z + f3 * w.w + sHopT[k][4];
            A[kc][h] = (_Float16)fmaxf(v, 0.0f);
        }
    }

    float ysum[8];
    #pragma unroll
    for (int r = 0; r < 8; ++r) ysum[r] = 0.0f;

    // 8 N-tiles x 4 K-chunks of v_wmma_f32_16x16x32_f16
    for (int t = 0; t < 8; ++t) {
        const int n = t * 16 + m;
        const _Float16* bp = &sW1[n * W1_STRIDE + (hi << 4)]; // 16B aligned
        v8f cf = {0.f, 0.f, 0.f, 0.f, 0.f, 0.f, 0.f, 0.f};
        #pragma unroll
        for (int kc = 0; kc < 4; ++kc) {
            // B fragment: lane holds column n, K = 32*kc + 16*hi + h
            union { v16h v; v8h part[2]; } B;
            const v8h* p = (const v8h*)(bp + 32 * kc);
            B.part[0] = p[0];
            B.part[1] = p[1];
            cf = __builtin_amdgcn_wmma_f32_16x16x32_f16(
                false, A[kc], false, B.v, (short)0, cf, false, false);
        }
        // C layout: VGPR r <-> row (r + 8*hi), col n. Fuse bias+ReLU+proj.
        const float2 bw = sBW[n];
        #pragma unroll
        for (int r = 0; r < 8; ++r)
            ysum[r] += fmaxf(cf[r] + bw.x, 0.0f) * bw.y;
    }

    // Reduce across the 16 lanes of each half-wave (rows stay separated).
    #pragma unroll
    for (int off = 8; off >= 1; off >>= 1) {
        #pragma unroll
        for (int r = 0; r < 8; ++r)
            ysum[r] += __shfl_xor(ysum[r], off, 32);
    }

    if (m == 0) {
        const float b2v = b2[0];
        #pragma unroll
        for (int r = 0; r < 8; ++r)
            out[tile + (hi << 3) + r] = fmaxf(ysum[r] + b2v, 0.0f);
    }
}

extern "C" void kernel_launch(void* const* d_in, const int* in_sizes, int n_in,
                              void* d_out, int out_size, void* d_ws, size_t ws_size,
                              hipStream_t stream) {
    const int N = in_sizes[0];           // x: (N,1)
    const int E = in_sizes[2];           // edge_attr: (E,)

    const float*     x    = (const float*)d_in[0];
    const long long* ei   = (const long long*)d_in[1];  // int64 (2,E)
    const float*     ea   = (const float*)d_in[2];
    const float*     W0   = (const float*)d_in[3];
    const float*     W1   = (const float*)d_in[4];
    const float*     W2   = (const float*)d_in[5];
    const float*     W3   = (const float*)d_in[6];
    const float*     bias = (const float*)d_in[7];
    const float*     w1   = (const float*)d_in[8];
    const float*     b1   = (const float*)d_in[9];
    const float*     w2   = (const float*)d_in[10];
    const float*     b2   = (const float*)d_in[11];
    float* out = (float*)d_out;

    float* ws  = (float*)d_ws;
    float* deg = ws;            // becomes dis in place
    float* h1  = ws + (size_t)N;
    float* h2  = ws + (size_t)2 * N;
    float* h3  = ws + (size_t)3 * N;

    const long long* row = ei;
    const long long* col = ei + E;

    const int eb = (E + 255) / 256;
    const int nb = (N + 255) / 256;

    zero_f32_kernel<<<(4 * N + 255) / 256, 256, 0, stream>>>(ws, 4 * N);

    // Prefer int32 packed-index path if workspace allows (saves ~130MB HBM
    // and lets the replayed index stream live in the 192MB L2).
    const size_t need32 = (size_t)4 * N * sizeof(float) + (size_t)2 * E * sizeof(int);
    if (ws_size >= need32) {
        int2* rc = (int2*)(ws + (size_t)4 * N);
        convert_deg_kernel<<<eb, 256, 0, stream>>>(row, col, ea, rc, deg, E);
        dis_kernel<<<nb, 256, 0, stream>>>(deg, N);
        prop32_kernel<<<eb, 256, 0, stream>>>(rc, ea, deg, x,  h1, E);
        prop32_kernel<<<eb, 256, 0, stream>>>(rc, ea, deg, h1, h2, E);
        prop32_kernel<<<eb, 256, 0, stream>>>(rc, ea, deg, h2, h3, E);
    } else {
        deg_kernel<<<eb, 256, 0, stream>>>(col, ea, deg, E);
        dis_kernel<<<nb, 256, 0, stream>>>(deg, N);
        prop64_kernel<<<eb, 256, 0, stream>>>(row, col, ea, deg, x,  h1, E);
        prop64_kernel<<<eb, 256, 0, stream>>>(row, col, ea, deg, h1, h2, E);
        prop64_kernel<<<eb, 256, 0, stream>>>(row, col, ea, deg, h2, h3, E);
    }

    node_mlp_kernel<<<(N + 127) / 128, 256, 0, stream>>>(
        x, h1, h2, h3, W0, W1, W2, W3, bias, w1, b1, w2, b2, out, N);
}
